// GCN__80942953660639
// MI455X (gfx1250) — matrix operand order repaired and checked by reading, not compiled, not run on hardware
//
#include <hip/hip_runtime.h>

#define NNODES 100000
#define KDIM1  1433
#define NDIM1  140
#define NDIM2  7
#define NT1    9                 // ceil(140/16) N-tiles for layer 1
#define KS1    45                // ceil(1433/32) K-steps for layer 1
#define KS2    5                 // ceil(140/32)  K-steps for layer 2

typedef __attribute__((ext_vector_type(16))) __bf16 v16bf;
typedef __attribute__((ext_vector_type(8)))  float  v8f;

// ---------------------------------------------------------------- utilities
__global__ void zero_kernel(float* __restrict__ p, size_t n) {
    size_t i = (size_t)blockIdx.x * blockDim.x + threadIdx.x;
    size_t stride = (size_t)gridDim.x * blockDim.x;
    for (; i < n; i += stride) p[i] = 0.0f;
}

__global__ void degree_kernel(const int* __restrict__ src, const int* __restrict__ dst,
                              float* __restrict__ degS, float* __restrict__ degD, int E) {
    int i = blockIdx.x * blockDim.x + threadIdx.x;
    if (i < E) {
        atomicAdd(&degS[src[i]], 1.0f);
        atomicAdd(&degD[dst[i]], 1.0f);
    }
}

__global__ void rsqrt_kernel(float* __restrict__ p, int n) {
    int i = blockIdx.x * blockDim.x + threadIdx.x;
    if (i < n) p[i] = rsqrtf(fmaxf(p[i], 1.0f));
}

// out[i] = relu(A[i] * normD[row] + bias[col]); out may alias A (elementwise)
template <int NC>
__global__ void postact_kernel(const float* __restrict__ A, const float* __restrict__ normD,
                               const float* __restrict__ bias, size_t total,
                               float* __restrict__ out) {
    size_t i = (size_t)blockIdx.x * blockDim.x + threadIdx.x;
    size_t stride = (size_t)gridDim.x * blockDim.x;
    for (; i < total; i += stride) {
        int r = (int)(i / NC);
        int c = (int)(i % NC);
        out[i] = fmaxf(A[i] * normD[r] + bias[c], 0.0f);
    }
}

// ---------------------------------------------------------------- weight pre-pack
// Pack W [Kdim x Ndim] (row-major f32) into bf16 WMMA B-fragments:
//   out[(((ks*ntiles + t)*32 + lane)*16 + j)] = W[ks*32 + (lane>>4)*8 + (j&7) + (j>>3)*16][t*16 + (lane&15)]
__global__ void packB_kernel(const float* __restrict__ W, __bf16* __restrict__ out,
                             int Kdim, int Ndim, int ntiles, int ksteps) {
    int idx = blockIdx.x * blockDim.x + threadIdx.x;
    int total = ksteps * ntiles * 32 * 16;
    if (idx >= total) return;
    int j    = idx & 15;
    int lane = (idx >> 4) & 31;
    int ft   = idx >> 9;             // fragment index = ks*ntiles + t
    int t    = ft % ntiles;
    int ks   = ft / ntiles;
    int hf = lane >> 4, li = lane & 15;
    int n = t * 16 + li;
    int k = ks * 32 + hf * 8 + (j & 7) + (j >> 3) * 16;
    out[idx] = (__bf16)((k < Kdim && n < Ndim) ? W[(size_t)k * Ndim + n] : 0.0f);
}

// ---------------------------------------------------------------- A-fragment helpers
// 16-bit A 16x32 layout: lane-half 0 -> K {0..7,16..23}, half 1 -> K {8..15,24..31}
__device__ __forceinline__ void loadA_fast(const float* __restrict__ xrow, int kBase, int hf,
                                           float* __restrict__ f) {
    const int k0 = kBase + hf * 8;
#pragma unroll
    for (int j = 0; j < 8; ++j) {
        f[j]     = xrow[k0 + j];
        f[8 + j] = xrow[k0 + 16 + j];
    }
}

__device__ __forceinline__ void loadA_guard(const float* __restrict__ xrow, int kBase, int hf,
                                            int Kdim, float* __restrict__ f) {
    const int k0 = kBase + hf * 8;
#pragma unroll
    for (int j = 0; j < 8; ++j) {
        int ka = k0 + j, kb = k0 + 16 + j;
        f[j]     = (ka < Kdim) ? xrow[ka] : 0.0f;
        f[8 + j] = (kb < Kdim) ? xrow[kb] : 0.0f;
    }
}

__device__ __forceinline__ v16bf cvtA(const float* __restrict__ f) {
    v16bf a;
#pragma unroll
    for (int j = 0; j < 16; ++j) a[j] = (__bf16)f[j];
    return a;
}

// ---------------------------------------------------------------- GEMM1: H1 = (X @ W0) * normS
// 256 threads = 8 waves; wave w owns rows [blk*128 + w*16, +16), all 9 N-tiles.
// Row index clamped (OOB rows compute garbage, never stored) -> no divergence in hot loop.
// A (raw f32) is loaded one K-step ahead; B is pipelined one tile ahead.
__global__ __launch_bounds__(256) void gemm1_kernel(const float* __restrict__ X,
                                                    const __bf16* __restrict__ Bp,
                                                    const float* __restrict__ normS,
                                                    float* __restrict__ H1) {
    const int tid  = threadIdx.x;
    const int wv   = tid >> 5;
    const int lane = tid & 31;
    const int hf   = lane >> 4;
    const int li   = lane & 15;
    const int mBase = blockIdx.x * 128 + wv * 16;
    const int arow  = min(mBase + li, NNODES - 1);          // clamp: safe, store-guarded
    const float* xrow = X + (size_t)arow * KDIM1;

    v8f acc[NT1];
    const v8f vzero = {0, 0, 0, 0, 0, 0, 0, 0};
#pragma unroll
    for (int t = 0; t < NT1; ++t) acc[t] = vzero;

    float fa[16];
    loadA_fast(xrow, 0, hf, fa);

    for (int ks = 0; ks < KS1; ++ks) {
        v16bf a = cvtA(fa);
        // prefetch next K-step's A while this step's WMMAs run (uniform branch)
        if (ks + 1 < KS1) {
            if (ks + 1 == KS1 - 1) loadA_guard(xrow, (ks + 1) * 32, hf, KDIM1, fa);
            else                   loadA_fast(xrow, (ks + 1) * 32, hf, fa);
        }

        const __bf16* bp = Bp + ((size_t)ks * NT1 * 32 + lane) * 16;
        v16bf bcur = *(const v16bf*)bp;
#pragma unroll
        for (int t = 0; t < NT1; ++t) {
            v16bf bnext = bcur;
            if (t + 1 < NT1) bnext = *(const v16bf*)(bp + (size_t)(t + 1) * (32 * 16));
            acc[t] = __builtin_amdgcn_wmma_f32_16x16x32_bf16(
                false, a, false, bcur, (short)0, acc[t], false, false);
            bcur = bnext;
        }
    }

    // store, scaled by src-norm. C/D layout: VGPR r -> row mBase + hf*8 + r, col = t*16 + li
    float ns[8];
#pragma unroll
    for (int r = 0; r < 8; ++r) {
        int rr = mBase + hf * 8 + r;
        ns[r] = (rr < NNODES) ? normS[rr] : 0.0f;
    }
#pragma unroll
    for (int t = 0; t < NT1; ++t) {
        int col = t * 16 + li;
        if (col < NDIM1) {
#pragma unroll
            for (int r = 0; r < 8; ++r) {
                int rr = mBase + hf * 8 + r;
                if (rr < NNODES) H1[(size_t)rr * NDIM1 + col] = acc[t][r] * ns[r];
            }
        }
    }
}

// ---------------------------------------------------------------- GEMM2: H2 = (P @ W1) * normS
__global__ __launch_bounds__(256) void gemm2_kernel(const float* __restrict__ P,
                                                    const __bf16* __restrict__ Bp,
                                                    const float* __restrict__ normS,
                                                    float* __restrict__ H2) {
    const int tid  = threadIdx.x;
    const int wv   = tid >> 5;
    const int lane = tid & 31;
    const int hf   = lane >> 4;
    const int li   = lane & 15;
    const int mBase = blockIdx.x * 128 + wv * 16;
    const int arow  = min(mBase + li, NNODES - 1);
    const float* prow = P + (size_t)arow * NDIM1;

    v8f acc = {0, 0, 0, 0, 0, 0, 0, 0};

    float fa[16];
    loadA_fast(prow, 0, hf, fa);
#pragma unroll
    for (int ks = 0; ks < KS2; ++ks) {
        v16bf a = cvtA(fa);
        if (ks + 1 < KS2) {
            if (ks + 1 == KS2 - 1) loadA_guard(prow, (ks + 1) * 32, hf, NDIM1, fa);
            else                   loadA_fast(prow, (ks + 1) * 32, hf, fa);
        }
        v16bf b = *(const v16bf*)(Bp + ((size_t)ks * 32 + lane) * 16);
        acc = __builtin_amdgcn_wmma_f32_16x16x32_bf16(
            false, a, false, b, (short)0, acc, false, false);
    }

    if (li < NDIM2) {
#pragma unroll
        for (int r = 0; r < 8; ++r) {
            int rr = mBase + hf * 8 + r;
            if (rr < NNODES) H2[(size_t)rr * NDIM2 + li] = acc[r] * normS[rr];
        }
    }
}

// ---------------------------------------------------------------- SpMM scatter-add
// One wave per edge; lanes sweep the 140 columns (coalesced, L2-resident atomics).
__global__ __launch_bounds__(256) void agg140_kernel(const int* __restrict__ src,
                                                     const int* __restrict__ dst,
                                                     const float* __restrict__ H,
                                                     float* __restrict__ AGG, int E) {
    int wv = threadIdx.x >> 5;
    int lane = threadIdx.x & 31;
    int e = blockIdx.x * 8 + wv;
    if (e >= E) return;
    int s = src[e], d = dst[e];
    const float* hs = H + (size_t)s * NDIM1;
    float* ad = AGG + (size_t)d * NDIM1;
    for (int c = lane; c < NDIM1; c += 32) atomicAdd(&ad[c], hs[c]);
}

// 8 threads per edge (7 active) for the 7-wide layer-2 aggregation.
__global__ void agg7_kernel(const int* __restrict__ src, const int* __restrict__ dst,
                            const float* __restrict__ H, float* __restrict__ AGG, int E) {
    size_t idx = (size_t)blockIdx.x * blockDim.x + threadIdx.x;
    int e = (int)(idx >> 3);
    int c = (int)(idx & 7);
    if (e < E && c < NDIM2)
        atomicAdd(&AGG[(size_t)dst[e] * NDIM2 + c], H[(size_t)src[e] * NDIM2 + c]);
}

// ---------------------------------------------------------------- launch
extern "C" void kernel_launch(void* const* d_in, const int* in_sizes, int n_in,
                              void* d_out, int out_size, void* d_ws, size_t ws_size,
                              hipStream_t stream) {
    const float* X   = (const float*)d_in[0];
    const int*   src = (const int*)d_in[1];
    const int*   dst = (const int*)d_in[2];
    const float* W0  = (const float*)d_in[3];
    const float* b0  = (const float*)d_in[4];
    const float* W1  = (const float*)d_in[5];
    const float* b1  = (const float*)d_in[6];
    float* out = (float*)d_out;
    const int E = in_sizes[1];

    // workspace: zeroed region first (normS|normD|AGG1|AGG2), then packed weights + H1|H2
    float* ws    = (float*)d_ws;
    float* normS = ws;                                       // [N] (deg -> rsqrt in place)
    float* normD = normS + NNODES;                           // [N]
    float* AGG1  = normD + NNODES;                           // [N*140]
    float* AGG2  = AGG1 + (size_t)NNODES * NDIM1;            // [N*7]
    float* after = AGG2 + (size_t)NNODES * NDIM2;            // 32B-aligned (offset % 8 == 0)
    __bf16* W0p  = (__bf16*)after;                           // [KS1*NT1*32*16] bf16
    const size_t w0pElems = (size_t)KS1 * NT1 * 32 * 16;
    __bf16* W1p  = W0p + w0pElems;                           // [KS2*1*32*16] bf16
    const size_t w1pElems = (size_t)KS2 * 1 * 32 * 16;
    float* H1    = (float*)(W1p + w1pElems);                 // [N*140]
    float* H2    = H1 + (size_t)NNODES * NDIM1;              // [N*7]

    const size_t zeroCount =
        (size_t)2 * NNODES + (size_t)NNODES * NDIM1 + (size_t)NNODES * NDIM2;
    zero_kernel<<<2048, 256, 0, stream>>>(ws, zeroCount);

    packB_kernel<<<(int)((w0pElems + 255) / 256), 256, 0, stream>>>(
        W0, W0p, KDIM1, NDIM1, NT1, KS1);
    packB_kernel<<<(int)((w1pElems + 255) / 256), 256, 0, stream>>>(
        W1, W1p, NDIM1, NDIM2, 1, KS2);

    degree_kernel<<<(E + 255) / 256, 256, 0, stream>>>(src, dst, normS, normD, E);
    rsqrt_kernel<<<(2 * NNODES + 255) / 256, 256, 0, stream>>>(ws, 2 * NNODES);

    const int gemmBlocks = (NNODES + 127) / 128;
    gemm1_kernel<<<gemmBlocks, 256, 0, stream>>>(X, W0p, normS, H1);

    agg140_kernel<<<(E + 7) / 8, 256, 0, stream>>>(src, dst, H1, AGG1, E);

    const size_t t1 = (size_t)NNODES * NDIM1;
    postact_kernel<NDIM1><<<4096, 256, 0, stream>>>(AGG1, normD, b0, t1, AGG1);

    gemm2_kernel<<<gemmBlocks, 256, 0, stream>>>(AGG1, W1p, normS, H2);

    const size_t a2threads = (size_t)E * 8;
    agg7_kernel<<<(unsigned)((a2threads + 255) / 256), 256, 0, stream>>>(src, dst, H2, AGG2, E);

    const size_t t2 = (size_t)NNODES * NDIM2;
    postact_kernel<NDIM2><<<1024, 256, 0, stream>>>(AGG2, normD, b1, t2, out);
}
